// GRUManual_78494822301753
// MI455X (gfx1250) — matrix operand order repaired and checked by reading, not compiled, not run on hardware
//
#include <hip/hip_runtime.h>

// GRU: B=256, T=2048, I=64, H=256  (MI455X / gfx1250, wave32, WMMA bf16)
//  kernel 1: pack combined weights [K=320][N=1024] (z, r, n_h, n_x groups) into
//            bf16 WMMA-B-fragment order; fold biases.
//  kernel 2: persistent scan, 16 WGs x 512 thr, each WG owns 16 batch rows for
//            all 2048 steps. h state in registers, bf16 A tile double-buffered
//            in LDS (1 barrier/step). Weights are streamed from L2 every step;
//            an opaque zero offset (asm barrier) per iteration prevents LICM
//            from hoisting 640 KB of fragments into spilled registers, while
//            keeping addrspace provenance so loads lower to global_load_b128
//            (LOADcnt only) instead of flat_load_b128 (LOADcnt + DScnt).

#define BB 256
#define TT 2048
#define II 64
#define HH 256

#define KTOT 320                    // H + I
#define NTOT 1024                   // 4 gate groups * H
#define KT 10                       // K tiles of 32
#define NT 64                       // N tiles of 16
#define NPACK (NT * KT * 512)       // bf16 elements in packed weights (327680)
#define WPK_BYTES (NPACK * 2)       // 655360
#define BC_OFF WPK_BYTES

typedef __attribute__((ext_vector_type(16))) __bf16 v16bf;
typedef __attribute__((ext_vector_type(8)))  float  v8f;

union FragU { uint4 u4[2]; v16bf v; };

static __device__ __forceinline__ unsigned short f2bf(float f) {
  unsigned int u = __float_as_uint(f);
  u += 0x7FFFu + ((u >> 16) & 1u);   // round-to-nearest-even
  return (unsigned short)(u >> 16);
}

// fast activations: v_exp_f32 + v_rcp_f32, no IEEE divide chain
static __device__ __forceinline__ float fsigmoid(float x) {
  return __builtin_amdgcn_rcpf(1.f + __expf(-x));
}
static __device__ __forceinline__ float ftanh(float x) {
  return 1.f - 2.f * __builtin_amdgcn_rcpf(1.f + __expf(2.f * x));
}

// ---------------------------------------------------------------------------
// Pack kernel: bf16 weight fragments + combined bias in workspace.
// B-fragment layout used here, contiguous k matching A (V_WMMA_F32_16X16X32_BF16, 32x16 B):
//   lane l<16 : column N = nt*16+l,      K = kt*32 + e       (e = 0..15)
//   lane l>=16: column N = nt*16+(l-16), K = kt*32 + 16 + e
// ---------------------------------------------------------------------------
__global__ void gru_pack(const float* __restrict__ Wiz, const float* __restrict__ Wir,
                         const float* __restrict__ Win, const float* __restrict__ Whz,
                         const float* __restrict__ Whr, const float* __restrict__ Whn,
                         const float* __restrict__ biz, const float* __restrict__ bir,
                         const float* __restrict__ bin, const float* __restrict__ bhz,
                         const float* __restrict__ bhr, const float* __restrict__ bhn,
                         unsigned short* __restrict__ wpk, float* __restrict__ bc)
{
  int gid = blockIdx.x * blockDim.x + threadIdx.x;
  if (gid < NPACK) {
    int frag = gid >> 9;
    int lane = (gid >> 4) & 31;
    int e    = gid & 15;
    int kt   = frag % KT;
    int nt   = frag / KT;
    int N    = nt * 16 + (lane & 15);
    int K    = kt * 32 + (lane >> 4) * 16 + e;
    int g    = N >> 8;     // 0=z, 1=r, 2=n_hidden, 3=n_input
    int j    = N & 255;
    float v = 0.f;
    if (K < HH) {
      if      (g == 0) v = Whz[j * HH + K];
      else if (g == 1) v = Whr[j * HH + K];
      else if (g == 2) v = Whn[j * HH + K];
    } else {
      int i = K - HH;
      if      (g == 0) v = Wiz[j * II + i];
      else if (g == 1) v = Wir[j * II + i];
      else if (g == 3) v = Win[j * II + i];
    }
    wpk[gid] = f2bf(v);
  } else if (gid < NPACK + NTOT) {
    int n = gid - NPACK;
    int g = n >> 8, j = n & 255;
    float v;
    if      (g == 0) v = bhz[j] + biz[j];
    else if (g == 1) v = bhr[j] + bir[j];
    else if (g == 2) v = bhn[j];
    else             v = bin[j];
    bc[n] = v;
  }
}

// ---------------------------------------------------------------------------
// Scan kernel
// ---------------------------------------------------------------------------
__global__ __launch_bounds__(512, 1)
void gru_scan(const float* __restrict__ u_seq,
              const unsigned short* __restrict__ wpk,
              const float* __restrict__ bc,
              const float* __restrict__ fcW,
              const float* __restrict__ fcb,
              float* __restrict__ y)
{
  __shared__ unsigned short Abuf[2][16][336]; // bf16 A: [0,256)=h, [256,320)=u_t
  __shared__ float yacc[2][16];

  const int tid  = threadIdx.x;
  const int lane = tid & 31;
  const int w    = tid >> 5;              // wave 0..15 -> hidden cols [w*16, w*16+16)
  const int b0   = blockIdx.x * 16;

  for (int idx = tid; idx < 2 * 16 * 336; idx += 512)
    ((unsigned short*)Abuf)[idx] = 0;
  if (tid < 32) ((float*)yacc)[tid] = 0.f;

  // per-lane constants
  const int   j     = w * 16 + (lane & 15);
  const float bz    = bc[j];
  const float br    = bc[256 + j];
  const float bnh   = bc[512 + j];
  const float bnx   = bc[768 + j];
  const float fw    = fcW[j];
  const float fb    = fcb[0];
  const int   mbase = (lane >> 4) * 8;
  const int   arow  = lane & 15;
  const int   ah8   = (lane >> 4) ? 16 : 0;

  // per-gate weight base pointers (kt offsets become immediate offsets)
  const char* bp0 = (const char*)wpk + (size_t)(0 * 16 + w) * (KT * 1024) + (size_t)lane * 32;
  const char* bp1 = (const char*)wpk + (size_t)(1 * 16 + w) * (KT * 1024) + (size_t)lane * 32;
  const char* bp2 = (const char*)wpk + (size_t)(2 * 16 + w) * (KT * 1024) + (size_t)lane * 32;
  const char* bp3 = (const char*)wpk + (size_t)(3 * 16 + w) * (KT * 1024) + (size_t)lane * 32;

  // u staging coords (2 f32 per thread, 16 rows x 64 cols)
  const int ur = tid >> 5;
  const int uc = (tid & 31) * 2;
  const float* uptr = &u_seq[((size_t)(b0 + ur)) * TT * II + uc];

  // h state in registers: rows mbase..mbase+7 of column j
  float hreg[8];
  #pragma unroll
  for (int v = 0; v < 8; ++v) hreg[v] = 0.f;

  // prologue: stage u_0 into Abuf[0]
  {
    const float2 uv = *(const float2*)(uptr /* t = 0 */);
    *(unsigned int*)&Abuf[0][ur][256 + uc] =
        (unsigned int)f2bf(uv.x) | ((unsigned int)f2bf(uv.y) << 16);
  }
  __syncthreads();

  for (int t = 0; t < TT; ++t) {
    const int cur = t & 1, nxt = cur ^ 1;

    // Opaque zero offset, fresh each step: the 80 fragment loads below cannot
    // be hoisted out of the t-loop (no 320-VGPR spill), yet the pointers keep
    // their global-addrspace provenance -> global_load_b128, LOADcnt only.
    unsigned int zoff = 0;
    asm volatile("" : "+v"(zoff));
    const char* q0 = bp0 + zoff;
    const char* q1 = bp1 + zoff;
    const char* q2 = bp2 + zoff;
    const char* q3 = bp3 + zoff;

    // ---- stage u_{t+1} into Abuf[nxt] (overlaps with GEMM below) ----
    if (t + 1 < TT) {
      const float2 uv = *(const float2*)(uptr + (size_t)(t + 1) * II);
      *(unsigned int*)&Abuf[nxt][ur][256 + uc] =
          (unsigned int)f2bf(uv.x) | ((unsigned int)f2bf(uv.y) << 16);
    }

    // ---- fused GEMM: [16 x 320] x [320 x 1024] on Abuf[cur] ----
    const unsigned short* arp = &Abuf[cur][arow][ah8];
    v8f acc0 = {}, acc1 = {}, acc2 = {}, acc3 = {};
    #pragma unroll
    for (int kt = 0; kt < KT; ++kt) {
      FragU a;
      a.u4[0] = *(const uint4*)(arp + kt * 32);       // ds_load_b128
      a.u4[1] = *(const uint4*)(arp + kt * 32 + 8);   // ds_load_b128

      FragU f0, f1, f2, f3;
      const uint4* p0 = (const uint4*)(q0 + kt * 1024);
      const uint4* p1 = (const uint4*)(q1 + kt * 1024);
      const uint4* p2 = (const uint4*)(q2 + kt * 1024);
      const uint4* p3 = (const uint4*)(q3 + kt * 1024);
      f0.u4[0] = p0[0]; f0.u4[1] = p0[1];             // global_load_b128 off:imm
      f1.u4[0] = p1[0]; f1.u4[1] = p1[1];
      f2.u4[0] = p2[0]; f2.u4[1] = p2[1];
      f3.u4[0] = p3[0]; f3.u4[1] = p3[1];

      acc0 = __builtin_amdgcn_wmma_f32_16x16x32_bf16(false, a.v, false, f0.v, (short)0, acc0, false, false);
      acc1 = __builtin_amdgcn_wmma_f32_16x16x32_bf16(false, a.v, false, f1.v, (short)0, acc1, false, false);
      acc2 = __builtin_amdgcn_wmma_f32_16x16x32_bf16(false, a.v, false, f2.v, (short)0, acc2, false, false);
      acc3 = __builtin_amdgcn_wmma_f32_16x16x32_bf16(false, a.v, false, f3.v, (short)0, acc3, false, false);
    }

    // ---- gates + state update (registers) + h_new -> Abuf[nxt] ----
    float yp[8];
    #pragma unroll
    for (int v = 0; v < 8; ++v) {
      const float zf = fsigmoid(acc0[v] + bz);
      const float rf = fsigmoid(acc1[v] + br);
      const float nf = ftanh(acc3[v] + bnx + rf * (acc2[v] + bnh));
      const float hn = (1.f - zf) * hreg[v] + zf * nf;
      hreg[v] = hn;
      Abuf[nxt][mbase + v][j] = f2bf(hn);
      yp[v] = hn * fw;
    }
    // fc reduction over this wave's 16 columns
    #pragma unroll
    for (int v = 0; v < 8; ++v) {
      float p = yp[v];
      p += __shfl_xor(p, 1);
      p += __shfl_xor(p, 2);
      p += __shfl_xor(p, 4);
      p += __shfl_xor(p, 8);
      if ((lane & 15) == 0) atomicAdd(&yacc[cur][mbase + v], p);  // ds_add_f32
    }

    __syncthreads();  // the only barrier per step

    if (tid < 16) {
      y[((size_t)(b0 + tid)) * TT + t] = yacc[cur][tid] + fb;
      yacc[cur][tid] = 0.f;
    }
  }
}

extern "C" void kernel_launch(void* const* d_in, const int* in_sizes, int n_in,
                              void* d_out, int out_size, void* d_ws, size_t ws_size,
                              hipStream_t stream)
{
  const float* u   = (const float*)d_in[0];
  const float* Wiz = (const float*)d_in[1];
  const float* biz = (const float*)d_in[2];
  const float* Wir = (const float*)d_in[3];
  const float* bir = (const float*)d_in[4];
  const float* Win = (const float*)d_in[5];
  const float* bin = (const float*)d_in[6];
  const float* Whz = (const float*)d_in[7];
  const float* bhz = (const float*)d_in[8];
  const float* Whr = (const float*)d_in[9];
  const float* bhr = (const float*)d_in[10];
  const float* Whn = (const float*)d_in[11];
  const float* bhn = (const float*)d_in[12];
  const float* fcW = (const float*)d_in[13];
  const float* fcb = (const float*)d_in[14];

  unsigned short* wpk = (unsigned short*)d_ws;
  float* bcv = (float*)((char*)d_ws + BC_OFF);

  const int total = NPACK + NTOT;
  gru_pack<<<(total + 255) / 256, 256, 0, stream>>>(
      Wiz, Wir, Win, Whz, Whr, Whn, biz, bir, bin, bhz, bhr, bhn, wpk, bcv);

  gru_scan<<<BB / 16, 512, 0, stream>>>(u, wpk, bcv, fcW, fcb, (float*)d_out);
}
